// MultiHeadDotProductAttention_23828478558762
// MI455X (gfx1250) — matrix operand order repaired and checked by reading, not compiled, not run on hardware
//
#include <hip/hip_runtime.h>

// ---------------------------------------------------------------------------
// Types for CDNA5 WMMA (wave32, 16x16x32 bf16 -> f32)
// ---------------------------------------------------------------------------
typedef __bf16 bf16;
typedef __attribute__((ext_vector_type(16))) __bf16 v16bf;
typedef __attribute__((ext_vector_type(8)))  __bf16 v8bf;
typedef __attribute__((ext_vector_type(8)))  float  v8f;

union Frag16 { v16bf v; v8bf h[2]; };

__device__ __forceinline__ v8f vzero8() {
    v8f z = {0.f, 0.f, 0.f, 0.f, 0.f, 0.f, 0.f, 0.f};
    return z;
}

__device__ __forceinline__ v8f wmma_bf16(v16bf a, v16bf b, v8f c) {
    return __builtin_amdgcn_wmma_f32_16x16x32_bf16(false, a, false, b,
                                                   (short)0, c, false, false);
}

// Async global->LDS copy, 16B per lane (ASYNCcnt-tracked).
__device__ __forceinline__ void async_copy_b128(const bf16* gaddr, unsigned lds_off) {
    asm volatile("global_load_async_to_lds_b128 %0, %1, off"
                 :: "v"(lds_off), "v"(gaddr)
                 : "memory");
}
__device__ __forceinline__ void wait_asynccnt0() {
    asm volatile("s_wait_asynccnt 0x0" ::: "memory");
}

// Problem constants
#define BB   2
#define SS   2048
#define DD   1024
#define HH   16
#define HDD  64

// ---------------------------------------------------------------------------
// Kernel 0: fp32 -> bf16 cast
// ---------------------------------------------------------------------------
__global__ void cvt_kernel(const float* __restrict__ in, bf16* __restrict__ out, int n) {
    int i = blockIdx.x * blockDim.x + threadIdx.x;
    int stride = gridDim.x * blockDim.x;
    for (; i < n; i += stride) out[i] = (bf16)in[i];
}

// ---------------------------------------------------------------------------
// GEMM kernels: block 128 thr = 4 waves; tile 64(M) x 128(N); wave 64x32.
// Double-buffered LDS: async A tile + vectorized/scattered transposed B tile
// for step k+1 issued before the WMMAs of step k.
// ---------------------------------------------------------------------------
// per-buffer LDS layout (elements): A[64][40] at 0, BT[128][40] at 2560
#define G_BT_OFF 2560
#define G_BUF_E  (2560 + 128 * 40)   // 7680 elements per buffer

__global__ void gemm_qkv_kernel(const bf16* __restrict__ xb,
                                const bf16* __restrict__ wqb,
                                const bf16* __restrict__ wkb,
                                const bf16* __restrict__ wvb,
                                const float* __restrict__ bq,
                                const float* __restrict__ bk,
                                const float* __restrict__ bv,
                                bf16* __restrict__ qout,
                                bf16* __restrict__ kout,
                                bf16* __restrict__ vout) {
    __shared__ __bf16 smem[2 * G_BUF_E];

    const int tid  = threadIdx.x;
    const int wave = tid >> 5;
    const int lane = tid & 31;
    const int l16  = lane & 15;
    const int half = lane >> 4;

    const int n0 = blockIdx.x * 128;
    const int m0 = blockIdx.y * 64;
    const int z  = blockIdx.z;

    const bf16*  wsel  = (z == 0) ? wqb : (z == 1) ? wkb : wvb;
    const float* bias  = (z == 0) ? bq  : (z == 1) ? bk  : bv;
    bf16*        outp  = (z == 0) ? qout : (z == 1) ? kout : vout;
    const float  scale = (z == 0) ? 0.125f : 1.0f;  // 1/sqrt(64)

    const int arow = tid >> 1, acol = (tid & 1) * 16;

    // stage K-tile k0 into buffer `buf`
    auto stage = [&](int k0, int buf) {
        const unsigned base = (unsigned)(buf * G_BUF_E);
        // A tile (64x32): async global->LDS, 32B per thread
        const bf16* src = xb + (size_t)(m0 + arow) * DD + k0 + acol;
        unsigned dst = (base + arow * 40 + acol) * 2u;
        async_copy_b128(src,     dst);
        async_copy_b128(src + 8, dst + 16);
        // B tile transposed (32x128 -> [n][k]): batch loads, then scatter
        v8bf w8[4];
#pragma unroll
        for (int i = 0; i < 4; ++i) {
            int id  = tid + 128 * i;
            int row = id >> 4;
            int cc  = (id & 15) * 8;
            w8[i] = *(const v8bf*)(wsel + (size_t)(k0 + row) * DD + n0 + cc);
        }
#pragma unroll
        for (int i = 0; i < 4; ++i) {
            int id  = tid + 128 * i;
            int row = id >> 4;
            int cc  = (id & 15) * 8;
#pragma unroll
            for (int j = 0; j < 8; ++j)
                smem[base + G_BT_OFF + (cc + j) * 40 + row] = w8[i][j];
        }
        if (k0 + 32 < DD) {  // prefetch the tile after next
            __builtin_prefetch(xb + (size_t)(m0 + arow) * DD + k0 + 32, 0, 1);
            __builtin_prefetch(wsel + (size_t)(k0 + 32 + (tid >> 4)) * DD + n0, 0, 1);
        }
    };

    v8f acc[4][2];
#pragma unroll
    for (int mt = 0; mt < 4; ++mt)
#pragma unroll
        for (int nt = 0; nt < 2; ++nt) acc[mt][nt] = vzero8();

    stage(0, 0);

    int cur = 0;
    for (int k0 = 0; k0 < DD; k0 += 32, cur ^= 1) {
        wait_asynccnt0();
        __syncthreads();
        if (k0 + 32 < DD) stage(k0 + 32, cur ^ 1);

        const __bf16* sb = smem + cur * G_BUF_E;
        Frag16 bf[2];
#pragma unroll
        for (int nt = 0; nt < 2; ++nt) {
            int nl = wave * 32 + nt * 16 + l16;
            const __bf16* pb = sb + G_BT_OFF + nl * 40 + half * 16;
            bf[nt].h[0] = *(const v8bf*)pb;
            bf[nt].h[1] = *(const v8bf*)(pb + 8);
        }
#pragma unroll
        for (int mt = 0; mt < 4; ++mt) {
            Frag16 af;
            const __bf16* pa = sb + (mt * 16 + l16) * 40 + half * 16;
            af.h[0] = *(const v8bf*)pa;
            af.h[1] = *(const v8bf*)(pa + 8);
#pragma unroll
            for (int nt = 0; nt < 2; ++nt)
                acc[mt][nt] = wmma_bf16(af.v, bf[nt].v, acc[mt][nt]);
        }
    }

    // ---- epilogue: scale, bias, write bf16 in [B,H,S,HD] layout ----
#pragma unroll
    for (int mt = 0; mt < 4; ++mt)
#pragma unroll
        for (int nt = 0; nt < 2; ++nt)
#pragma unroll
            for (int e = 0; e < 8; ++e) {
                int row = m0 + mt * 16 + 8 * half + e;
                int col = n0 + wave * 32 + nt * 16 + l16;
                float val = acc[mt][nt][e] * scale + bias[col];
                int b_ = row >> 11, s_ = row & (SS - 1);
                int h_ = col >> 6,  hd = col & (HDD - 1);
                outp[(((size_t)(b_ * HH + h_) * SS + s_) << 6) + hd] = (bf16)val;
            }
}

// ---------------------------------------------------------------------------
// Kernel 2: flash attention per (b,h). Block 128 thr = 4 waves.
// Double-buffered K/V chunk staging (async K copy, transposed V scatter).
// ---------------------------------------------------------------------------
// per-buffer (elements): K[32][64] at 0, VT[64][40] at 2048 -> 4608/buffer
// P[4][32][40] at 9216
#define A_VT_OFF 2048
#define A_BUF_E  4608
#define A_P_OFF  9216
#define A_SMEM_E (9216 + 4 * 32 * 40)

__global__ void attn_kernel(const bf16* __restrict__ qb,
                            const bf16* __restrict__ kb,
                            const bf16* __restrict__ vb,
                            bf16* __restrict__ ctxb) {
    __shared__ __bf16 smem[A_SMEM_E];

    const int tid  = threadIdx.x;
    const int wave = tid >> 5;
    const int lane = tid & 31;
    const int l16  = lane & 15;
    const int half = lane >> 4;

    const int bh = blockIdx.y;
    const int b_ = bh >> 4, h_ = bh & 15;
    const int q0 = blockIdx.x * 128 + wave * 32;

    const bf16* qbase = qb + (size_t)bh * SS * HDD;
    const bf16* kbase = kb + (size_t)bh * SS * HDD;
    const bf16* vbase = vb + (size_t)bh * SS * HDD;

    const int krow = tid >> 2, kcol = (tid & 3) * 16;

    auto stage = [&](int kv0, int buf) {
        const unsigned base = (unsigned)(buf * A_BUF_E);
        // K chunk (32x64): async global->LDS
        const bf16* src = kbase + (size_t)(kv0 + krow) * HDD + kcol;
        unsigned dst = (base + krow * 64 + kcol) * 2u;
        async_copy_b128(src,     dst);
        async_copy_b128(src + 8, dst + 16);
        // V chunk transposed -> [hd][kv]: batch loads, then scatter
        v8bf v8[2];
#pragma unroll
        for (int i = 0; i < 2; ++i) {
            int id = tid + 128 * i;
            int kv = id >> 3;
            int hd = (id & 7) * 8;
            v8[i] = *(const v8bf*)(vbase + (size_t)(kv0 + kv) * HDD + hd);
        }
#pragma unroll
        for (int i = 0; i < 2; ++i) {
            int id = tid + 128 * i;
            int kv = id >> 3;
            int hd = (id & 7) * 8;
#pragma unroll
            for (int j = 0; j < 8; ++j)
                smem[base + A_VT_OFF + (hd + j) * 40 + kv] = v8[i][j];
        }
        if (kv0 + 32 < SS) {
            __builtin_prefetch(kbase + (size_t)(kv0 + 32 + krow) * HDD, 0, 1);
            __builtin_prefetch(vbase + (size_t)(kv0 + 32 + (tid >> 3)) * HDD, 0, 1);
        }
    };

    // preload Q fragments
    Frag16 aq[2][2];
#pragma unroll
    for (int mt = 0; mt < 2; ++mt)
#pragma unroll
        for (int kh = 0; kh < 2; ++kh) {
            int srow = q0 + mt * 16 + l16;
            const bf16* p = qbase + (size_t)srow * HDD + kh * 32 + half * 16;
            aq[mt][kh].h[0] = *(const v8bf*)p;
            aq[mt][kh].h[1] = *(const v8bf*)(p + 8);
        }

    v8f ctx[2][4];
    float m_run[2][8], l_run[2][8];
#pragma unroll
    for (int mt = 0; mt < 2; ++mt) {
#pragma unroll
        for (int nh = 0; nh < 4; ++nh) ctx[mt][nh] = vzero8();
#pragma unroll
        for (int e = 0; e < 8; ++e) { m_run[mt][e] = -1e30f; l_run[mt][e] = 0.f; }
    }

    stage(0, 0);

    int cur = 0;
    for (int kv0 = 0; kv0 < SS; kv0 += 32, cur ^= 1) {
        wait_asynccnt0();
        __syncthreads();
        if (kv0 + 32 < SS) stage(kv0 + 32, cur ^ 1);

        const __bf16* sk = smem + cur * A_BUF_E;         // K chunk
        const __bf16* sv = sk + A_VT_OFF;                // VT chunk

        // ---- scores ----
        v8f s[2][2];
#pragma unroll
        for (int mt = 0; mt < 2; ++mt)
#pragma unroll
            for (int nt = 0; nt < 2; ++nt) {
                s[mt][nt] = vzero8();
#pragma unroll
                for (int kh = 0; kh < 2; ++kh) {
                    Frag16 bk_;
                    const __bf16* p = sk + (nt * 16 + l16) * 64 + kh * 32 + half * 16;
                    bk_.h[0] = *(const v8bf*)p;
                    bk_.h[1] = *(const v8bf*)(p + 8);
                    s[mt][nt] = wmma_bf16(aq[mt][kh].v, bk_.v, s[mt][nt]);
                }
            }

        // ---- online softmax + ctx accumulation ----
#pragma unroll
        for (int mt = 0; mt < 2; ++mt) {
            float alpha[8];
#pragma unroll
            for (int e = 0; e < 8; ++e) {
                float cm = fmaxf(s[mt][0][e], s[mt][1][e]);
#pragma unroll
                for (int off = 1; off < 16; off <<= 1)
                    cm = fmaxf(cm, __shfl_xor(cm, off, 32));
                float mo = m_run[mt][e];
                float mn = fmaxf(mo, cm);
                float a_ = __expf(mo - mn);
                float p0 = __expf(s[mt][0][e] - mn);
                float p1 = __expf(s[mt][1][e] - mn);
                s[mt][0][e] = p0; s[mt][1][e] = p1;
                float rs = p0 + p1;
#pragma unroll
                for (int off = 1; off < 16; off <<= 1)
                    rs += __shfl_xor(rs, off, 32);
                l_run[mt][e] = l_run[mt][e] * a_ + rs;
                m_run[mt][e] = mn;
                alpha[e] = a_;
            }
#pragma unroll
            for (int nh = 0; nh < 4; ++nh)
#pragma unroll
                for (int e = 0; e < 8; ++e) ctx[mt][nh][e] *= alpha[e];

            // P: C-layout -> per-wave LDS (bf16)
#pragma unroll
            for (int nt = 0; nt < 2; ++nt)
#pragma unroll
                for (int e = 0; e < 8; ++e)
                    smem[A_P_OFF + wave * 1280 +
                         (mt * 16 + 8 * half + e) * 40 + nt * 16 + l16] =
                        (bf16)s[mt][nt][e];

            // reload P as A-fragment; WMMA against transposed V
            Frag16 pa;
            {
                const __bf16* p =
                    smem + A_P_OFF + wave * 1280 + (mt * 16 + l16) * 40 + half * 16;
                pa.h[0] = *(const v8bf*)p;
                pa.h[1] = *(const v8bf*)(p + 8);
            }
#pragma unroll
            for (int nh = 0; nh < 4; ++nh) {
                Frag16 bv_;
                const __bf16* p = sv + (nh * 16 + l16) * 40 + half * 16;
                bv_.h[0] = *(const v8bf*)p;
                bv_.h[1] = *(const v8bf*)(p + 8);
                ctx[mt][nh] = wmma_bf16(pa.v, bv_.v, ctx[mt][nh]);
            }
        }
    }

    // ---- finalize ----
#pragma unroll
    for (int mt = 0; mt < 2; ++mt)
#pragma unroll
        for (int nh = 0; nh < 4; ++nh)
#pragma unroll
            for (int e = 0; e < 8; ++e) {
                float val = ctx[mt][nh][e] / l_run[mt][e];
                int srow = q0 + mt * 16 + 8 * half + e;
                int hd   = nh * 16 + l16;
                ctxb[((size_t)(b_ * SS + srow)) * DD + h_ * HDD + hd] = (bf16)val;
            }
}

// ---------------------------------------------------------------------------
// Kernel 3: output projection  out[4096,1024] = ctx_bf @ Wo_bf + bo (f32 out)
// ---------------------------------------------------------------------------
__global__ void gemm_out_kernel(const bf16* __restrict__ ab,
                                const bf16* __restrict__ wb,
                                const float* __restrict__ bo,
                                float* __restrict__ outp) {
    __shared__ __bf16 smem[2 * G_BUF_E];

    const int tid  = threadIdx.x;
    const int wave = tid >> 5;
    const int lane = tid & 31;
    const int l16  = lane & 15;
    const int half = lane >> 4;

    const int n0 = blockIdx.x * 128;
    const int m0 = blockIdx.y * 64;

    const int arow = tid >> 1, acol = (tid & 1) * 16;

    auto stage = [&](int k0, int buf) {
        const unsigned base = (unsigned)(buf * G_BUF_E);
        const bf16* src = ab + (size_t)(m0 + arow) * DD + k0 + acol;
        unsigned dst = (base + arow * 40 + acol) * 2u;
        async_copy_b128(src,     dst);
        async_copy_b128(src + 8, dst + 16);
        v8bf w8[4];
#pragma unroll
        for (int i = 0; i < 4; ++i) {
            int id  = tid + 128 * i;
            int row = id >> 4;
            int cc  = (id & 15) * 8;
            w8[i] = *(const v8bf*)(wb + (size_t)(k0 + row) * DD + n0 + cc);
        }
#pragma unroll
        for (int i = 0; i < 4; ++i) {
            int id  = tid + 128 * i;
            int row = id >> 4;
            int cc  = (id & 15) * 8;
#pragma unroll
            for (int j = 0; j < 8; ++j)
                smem[base + G_BT_OFF + (cc + j) * 40 + row] = w8[i][j];
        }
        if (k0 + 32 < DD) {
            __builtin_prefetch(ab + (size_t)(m0 + arow) * DD + k0 + 32, 0, 1);
            __builtin_prefetch(wb + (size_t)(k0 + 32 + (tid >> 4)) * DD + n0, 0, 1);
        }
    };

    v8f acc[4][2];
#pragma unroll
    for (int mt = 0; mt < 4; ++mt)
#pragma unroll
        for (int nt = 0; nt < 2; ++nt) acc[mt][nt] = vzero8();

    stage(0, 0);

    int cur = 0;
    for (int k0 = 0; k0 < DD; k0 += 32, cur ^= 1) {
        wait_asynccnt0();
        __syncthreads();
        if (k0 + 32 < DD) stage(k0 + 32, cur ^ 1);

        const __bf16* sb = smem + cur * G_BUF_E;
        Frag16 bf[2];
#pragma unroll
        for (int nt = 0; nt < 2; ++nt) {
            int nl = wave * 32 + nt * 16 + l16;
            const __bf16* pb = sb + G_BT_OFF + nl * 40 + half * 16;
            bf[nt].h[0] = *(const v8bf*)pb;
            bf[nt].h[1] = *(const v8bf*)(pb + 8);
        }
#pragma unroll
        for (int mt = 0; mt < 4; ++mt) {
            Frag16 af;
            const __bf16* pa = sb + (mt * 16 + l16) * 40 + half * 16;
            af.h[0] = *(const v8bf*)pa;
            af.h[1] = *(const v8bf*)(pa + 8);
#pragma unroll
            for (int nt = 0; nt < 2; ++nt)
                acc[mt][nt] = wmma_bf16(af.v, bf[nt].v, acc[mt][nt]);
        }
    }

#pragma unroll
    for (int mt = 0; mt < 4; ++mt)
#pragma unroll
        for (int nt = 0; nt < 2; ++nt)
#pragma unroll
            for (int e = 0; e < 8; ++e) {
                int row = m0 + mt * 16 + 8 * half + e;
                int col = n0 + wave * 32 + nt * 16 + l16;
                outp[(size_t)row * DD + col] = acc[mt][nt][e] + bo[col];
            }
}

// ---------------------------------------------------------------------------
// Host-side launcher
// ---------------------------------------------------------------------------
extern "C" void kernel_launch(void* const* d_in, const int* in_sizes, int n_in,
                              void* d_out, int out_size, void* d_ws, size_t ws_size,
                              hipStream_t stream) {
    const float* x  = (const float*)d_in[0];
    const float* Wq = (const float*)d_in[1];
    const float* bq = (const float*)d_in[2];
    const float* Wk = (const float*)d_in[3];
    const float* bk = (const float*)d_in[4];
    const float* Wv = (const float*)d_in[5];
    const float* bv = (const float*)d_in[6];
    const float* Wo = (const float*)d_in[7];
    const float* bo = (const float*)d_in[8];
    float* out = (float*)d_out;

    const size_t NX = (size_t)BB * SS * DD;
    const size_t NW = (size_t)DD * DD;

    bf16* xb   = (bf16*)d_ws;
    bf16* wqb  = xb  + NX;
    bf16* wkb  = wqb + NW;
    bf16* wvb  = wkb + NW;
    bf16* wob  = wvb + NW;
    bf16* qh   = wob + NW;
    bf16* kh   = qh  + NX;
    bf16* vh   = kh  + NX;
    bf16* ctxb = vh  + NX;

    cvt_kernel<<<dim3(2048), dim3(256), 0, stream>>>(x,  xb,  (int)NX);
    cvt_kernel<<<dim3(1024), dim3(256), 0, stream>>>(Wq, wqb, (int)NW);
    cvt_kernel<<<dim3(1024), dim3(256), 0, stream>>>(Wk, wkb, (int)NW);
    cvt_kernel<<<dim3(1024), dim3(256), 0, stream>>>(Wv, wvb, (int)NW);
    cvt_kernel<<<dim3(1024), dim3(256), 0, stream>>>(Wo, wob, (int)NW);

    gemm_qkv_kernel<<<dim3(DD / 128, (BB * SS) / 64, 3), dim3(128), 0, stream>>>(
        xb, wqb, wkb, wvb, bq, bk, bv, qh, kh, vh);

    attn_kernel<<<dim3(SS / 128, BB * HH), dim3(128), 0, stream>>>(qh, kh, vh, ctxb);

    gemm_out_kernel<<<dim3(DD / 128, (BB * SS) / 64), dim3(128), 0, stream>>>(
        ctxb, wob, bo, out);
}
